// MMTAs_90194313216639
// MI455X (gfx1250) — matrix operand order repaired and checked by reading, not compiled, not run on hardware
//
#include <hip/hip_runtime.h>

// ---------------------------------------------------------------------------
// MI455X (gfx1250) fused agent-attention.
// Shapes: B=2048, M=16, S=3, D=128, H=8, E=1024.
// Design: bf16 WMMA (16x16x32) projections packed 16-b-rows per tile, fully
// fused per (b,m) using the rank-1 structure of qa @ (ka @ vp). Only the
// transposed bf16 weights (12.6 MB, L2-resident) live in workspace.
// HBM traffic ~230 MB => ~10 us roofline at 23.3 TB/s.
// Round-2 changes: per-tile __syncthreads() to keep A-fragments from being
// hoisted (caps VGPRs < 256, more resident waves for L2-latency hiding), and
// async global->LDS staging of the fp32 residual overlapped with pass 2.
// ---------------------------------------------------------------------------

namespace {
constexpr int kB = 2048;
constexpr int kM = 16;
constexpr int kS = 3;
constexpr int kD = 128;
constexpr int kH = 8;
constexpr int kE = kH * kD; // 1024
constexpr float kLnEps = 1e-5f;
} // namespace

typedef __attribute__((ext_vector_type(8)))  unsigned short u16x8;
typedef __attribute__((ext_vector_type(16))) __bf16         v16bf;
typedef __attribute__((ext_vector_type(8)))  float          v8f;

static __device__ __forceinline__ unsigned short f32_to_bf16(float f) {
  unsigned int u = __float_as_uint(f);
  u += 0x7fffu + ((u >> 16) & 1u); // round-to-nearest-even
  return (unsigned short)(u >> 16);
}

static __device__ __forceinline__ v16bf load_frag(const unsigned short* p0,
                                                  const unsigned short* p1) {
  union { u16x8 h[2]; v16bf v; } u;
  u.h[0] = *(const u16x8*)p0;
  u.h[1] = *(const u16x8*)p1;
  return u.v;
}

// ---------------------------------------------------------------------------
// W[m][d][e] (f32) -> WT[m][e][d] (bf16), LDS-tiled so both sides coalesce.
// ---------------------------------------------------------------------------
__global__ void __launch_bounds__(256) convert_w_kernel(
    const float* __restrict__ W, unsigned short* __restrict__ WT) {
  __shared__ float tile[32][33];
  const int m  = blockIdx.z;
  const int e0 = blockIdx.x * 32;
  const int d0 = blockIdx.y * 32;
  const int tx = threadIdx.x; // 32
  const int ty = threadIdx.y; // 8
  const float*    Wm  = W  + (size_t)m * kD * kE;
  unsigned short* WTm = WT + (size_t)m * kE * kD;
#pragma unroll
  for (int r = ty; r < 32; r += 8)
    tile[r][tx] = Wm[(size_t)(d0 + r) * kE + (e0 + tx)];
  __syncthreads();
#pragma unroll
  for (int r = ty; r < 32; r += 8)
    WTm[(size_t)(e0 + r) * kD + (d0 + tx)] = f32_to_bf16(tile[tx][r]);
}

// ---------------------------------------------------------------------------
// 3 WMMA row-GEMMs (one per s) over one K=128 strip against one 16-col strip
// of WT.  A-fragments come from LDS (bf16 staged inputs), B-fragments from
// global bf16 transposed weights (L2-resident).
// ---------------------------------------------------------------------------
static __device__ __forceinline__ void gemm3(
    const unsigned short* a0, const unsigned short* a1,
    const unsigned short* a2, const unsigned short* __restrict__ WT,
    size_t bbase, v8f& c0, v8f& c1, v8f& c2) {
#pragma unroll
  for (int ks = 0; ks < 4; ++ks) {
    const int ko = ks * 32;
    const v16bf bf = load_frag(WT + bbase + ko, WT + bbase + ko + 16);
    const v16bf f0 = load_frag(a0 + ko, a0 + ko + 16);
    c0 = __builtin_amdgcn_wmma_f32_16x16x32_bf16(false, f0, false, bf,
                                                 (short)0, c0, false, false);
    const v16bf f1 = load_frag(a1 + ko, a1 + ko + 16);
    c1 = __builtin_amdgcn_wmma_f32_16x16x32_bf16(false, f1, false, bf,
                                                 (short)0, c1, false, false);
    const v16bf f2 = load_frag(a2 + ko, a2 + ko + 16);
    c2 = __builtin_amdgcn_wmma_f32_16x16x32_bf16(false, f2, false, bf,
                                                 (short)0, c2, false, false);
  }
}

// ---------------------------------------------------------------------------
// One block = (m, 16 consecutive b).  8 waves, each owns a 16-wide column
// strip of every 128-wide e-tile (8 tiles cover E=1024).
// ---------------------------------------------------------------------------
__global__ void __launch_bounds__(256) fused_agent_attn_kernel(
    const float* __restrict__ qg, const float* __restrict__ kg,
    const float* __restrict__ vg,
    const unsigned short* __restrict__ WqT,
    const unsigned short* __restrict__ WkT,
    const unsigned short* __restrict__ WvT,
    const float* __restrict__ ln_q_w, const float* __restrict__ ln_q_b,
    const float* __restrict__ ln_k_w, const float* __restrict__ ln_k_b,
    const float* __restrict__ bias1, const float* __restrict__ bias2,
    float* __restrict__ out) {
  __shared__ unsigned short s_x[3][kS][16][kD]; // bf16 staged q/k/v rows, 36KB
  __shared__ float s_qa[kS][16];
  __shared__ float s_ka[kS][16];
  __shared__ float s_wbar[16][kD];              // 8KB

  const int m    = blockIdx.y;
  const int b0   = blockIdx.x * 16;
  const int wave = threadIdx.x >> 5;
  const int lane = threadIdx.x & 31;
  const int l15  = lane & 15;
  const int hi   = lane >> 4;

  // ---- zero accumulators + stage inputs as bf16 into LDS ----
  for (int i = threadIdx.x; i < 16 * kD; i += 256) (&s_wbar[0][0])[i] = 0.f;
  if (threadIdx.x < kS * 16) {
    (&s_qa[0][0])[threadIdx.x] = 0.f;
    (&s_ka[0][0])[threadIdx.x] = 0.f;
  }
  for (int i = threadIdx.x; i < 3 * kS * 16 * kD; i += 256) {
    const int d  = i & (kD - 1);
    const int r  = i >> 7;      // 0..143
    const int bl = r & 15;
    const int r2 = r >> 4;      // 0..8
    const int s  = r2 % kS;
    const int x  = r2 / kS;
    const size_t g =
        (((size_t)(b0 + bl) * kM + m) * kS + s) * kD + d;
    const float* src = (x == 0) ? qg : ((x == 1) ? kg : vg);
    s_x[x][s][bl][d] = f32_to_bf16(src[g]);
  }
  __syncthreads();

  // A-fragment base pointers for this lane (16-bit A layout:
  // lanes<16 hold K=0..7 & 16..23, lanes>=16 hold K=8..15 & 24..31).
  const unsigned short* aq[kS];
  const unsigned short* ak[kS];
  const unsigned short* av[kS];
#pragma unroll
  for (int s = 0; s < kS; ++s) {
    aq[s] = &s_x[0][s][l15][hi * 8];
    ak[s] = &s_x[1][s][l15][hi * 8];
    av[s] = &s_x[2][s][l15][hi * 8];
  }

  // ---- Pass 1: qp & kp tiles, agent fragment, qa/ka partial dots ----
  for (int t = 0; t < 8; ++t) {
    // LDS clobber: stops A-fragment hoisting across tiles (caps VGPRs) and
    // keeps the 8 waves' B-strip loads clustered on the same 128-wide W tile.
    __syncthreads();
    const int ecol = t * 128 + wave * 16 + l15;
    // B layout: lanes<16 K=0..7 & 16..23, lanes>=16 K=8..15 & 24..31.
    const size_t bbase = ((size_t)m * kE + ecol) * kD + hi * 8;
    if (t < 7) {
      __builtin_prefetch((const void*)(WqT + bbase + 128 * kD), 0, 1);
      __builtin_prefetch((const void*)(WkT + bbase + 128 * kD), 0, 1);
    }

    v8f cq0 = {}, cq1 = {}, cq2 = {};
    gemm3(aq[0], aq[1], aq[2], WqT, bbase, cq0, cq1, cq2);

    v8f agent; // A = mean over s of qp, same fragment layout as C/D
#pragma unroll
    for (int i = 0; i < 8; ++i)
      agent[i] = (cq0[i] + cq1[i] + cq2[i]) * (1.f / 3.f);

#pragma unroll
    for (int i = 0; i < 8; ++i) { // C/D layout: row b = i + hi*8, col = l15
      const int bl = i + hi * 8;
      atomicAdd(&s_qa[0][bl], cq0[i] * agent[i]);
      atomicAdd(&s_qa[1][bl], cq1[i] * agent[i]);
      atomicAdd(&s_qa[2][bl], cq2[i] * agent[i]);
    }

    v8f ck0 = {}, ck1 = {}, ck2 = {};
    gemm3(ak[0], ak[1], ak[2], WkT, bbase, ck0, ck1, ck2);
#pragma unroll
    for (int i = 0; i < 8; ++i) {
      const int bl = i + hi * 8;
      atomicAdd(&s_ka[0][bl], ck0[i] * agent[i]);
      atomicAdd(&s_ka[1][bl], ck1[i] * agent[i]);
      atomicAdd(&s_ka[2][bl], ck2[i] * agent[i]);
    }
  }
  __syncthreads();

  // ---- tiny LN(3) + bias + softmax(3) per b, for qa and ka ----
  if (threadIdx.x < 16) {
    const int bl = threadIdx.x;
    {
      float x0 = s_qa[0][bl], x1 = s_qa[1][bl], x2 = s_qa[2][bl];
      float mu = (x0 + x1 + x2) * (1.f / 3.f);
      float d0 = x0 - mu, d1 = x1 - mu, d2 = x2 - mu;
      float inv = rsqrtf((d0 * d0 + d1 * d1 + d2 * d2) * (1.f / 3.f) + kLnEps);
      float y0 = d0 * inv * ln_q_w[0] + ln_q_b[0] + bias2[m * kS + 0];
      float y1 = d1 * inv * ln_q_w[1] + ln_q_b[1] + bias2[m * kS + 1];
      float y2 = d2 * inv * ln_q_w[2] + ln_q_b[2] + bias2[m * kS + 2];
      float mx = fmaxf(y0, fmaxf(y1, y2));
      float e0 = __expf(y0 - mx), e1 = __expf(y1 - mx), e2 = __expf(y2 - mx);
      float r = 1.f / (e0 + e1 + e2);
      s_qa[0][bl] = e0 * r; s_qa[1][bl] = e1 * r; s_qa[2][bl] = e2 * r;
    }
    {
      float x0 = s_ka[0][bl], x1 = s_ka[1][bl], x2 = s_ka[2][bl];
      float mu = (x0 + x1 + x2) * (1.f / 3.f);
      float d0 = x0 - mu, d1 = x1 - mu, d2 = x2 - mu;
      float inv = rsqrtf((d0 * d0 + d1 * d1 + d2 * d2) * (1.f / 3.f) + kLnEps);
      float y0 = d0 * inv * ln_k_w[0] + ln_k_b[0] + bias1[m * kS + 0];
      float y1 = d1 * inv * ln_k_w[1] + ln_k_b[1] + bias1[m * kS + 1];
      float y2 = d2 * inv * ln_k_w[2] + ln_k_b[2] + bias1[m * kS + 2];
      float mx = fmaxf(y0, fmaxf(y1, y2));
      float e0 = __expf(y0 - mx), e1 = __expf(y1 - mx), e2 = __expf(y2 - mx);
      float r = 1.f / (e0 + e1 + e2);
      s_ka[0][bl] = e0 * r; s_ka[1][bl] = e1 * r; s_ka[2][bl] = e2 * r;
    }
  }
  __syncthreads();

  // ---- async-stage the fp32 residual q rows into the dead q/k LDS space ----
  // s_x[0..1] (24KB bf16 staging) is dead after pass 1; overlay 3*16*128 f32.
  // ASYNCcnt-tracked DMA overlaps the whole of pass 2.
  {
    const unsigned lds_base = (unsigned)(size_t)(const void*)&s_x[0][0][0][0];
#pragma unroll
    for (int j = 0; j < 6; ++j) {
      const int f  = (wave * 6 + j) * 32 + lane; // float4 index, 0..1535
      const int d4 = f & 31;
      const int bl = (f >> 5) & 15;
      const int s  = f >> 9;
      const float* gp =
          qg + ((((size_t)(b0 + bl) * kM + m) * kS + s) * kD + d4 * 4);
      const unsigned lp = lds_base + (unsigned)f * 16u;
      asm volatile("global_load_async_to_lds_b128 %0, %1, off"
                   :: "v"(lp), "v"(gp) : "memory");
    }
  }

  // hoist this lane's ka rows (b = i + hi*8) out of the pass-2 loop
  float kr0[8], kr1[8], kr2[8];
#pragma unroll
  for (int i = 0; i < 8; ++i) {
    const int bl = i + hi * 8;
    kr0[i] = s_ka[0][bl]; kr1[i] = s_ka[1][bl]; kr2[i] = s_ka[2][bl];
  }

  // ---- Pass 2: vp tiles folded through ka into wbar (1/H included) ----
  for (int t = 0; t < 8; ++t) {
    __syncthreads(); // same anti-hoist / wave-clustering barrier as pass 1
    const int ecol  = t * 128 + wave * 16 + l15;
    const size_t bbase = ((size_t)m * kE + ecol) * kD + hi * 8;
    if (t < 7)
      __builtin_prefetch((const void*)(WvT + bbase + 128 * kD), 0, 1);

    v8f cv0 = {}, cv1 = {}, cv2 = {};
    gemm3(av[0], av[1], av[2], WvT, bbase, cv0, cv1, cv2);

    const int d = wave * 16 + l15; // h = t, folded by the sum over t
#pragma unroll
    for (int i = 0; i < 8; ++i) {
      const int bl = i + hi * 8;
      const float wv =
          (kr0[i] * cv0[i] + kr1[i] * cv1[i] + kr2[i] * cv2[i]) * (1.f / kH);
      atomicAdd(&s_wbar[bl][d], wv);
    }
  }

  // drain this wave's async DMA, then make all waves' LDS writes visible
  asm volatile("s_wait_asynccnt 0x0" ::: "memory");
  __syncthreads();

  // ---- rank-1 expand + residual (residual now in LDS) ----
  const float* s_res = (const float*)&s_x[0][0][0][0]; // [s][bl][d]
  for (int i = threadIdx.x; i < 16 * kS * kD; i += 256) {
    const int d  = i & (kD - 1);
    const int s  = (i >> 7) % kS;
    const int bl = i / (kS * kD);
    const size_t g = (((size_t)(b0 + bl) * kM + m) * kS + s) * kD + d;
    out[g] = s_qa[s][bl] * s_wbar[bl][d] + s_res[((s * 16 + bl) << 7) + d];
  }
}

// ---------------------------------------------------------------------------
extern "C" void kernel_launch(void* const* d_in, const int* in_sizes, int n_in,
                              void* d_out, int out_size, void* d_ws,
                              size_t ws_size, hipStream_t stream) {
  (void)in_sizes; (void)n_in; (void)out_size; (void)ws_size;

  const float* q      = (const float*)d_in[0];
  const float* k      = (const float*)d_in[1];
  const float* v      = (const float*)d_in[2];
  const float* Wq     = (const float*)d_in[3];
  const float* Wk     = (const float*)d_in[4];
  const float* Wv     = (const float*)d_in[5];
  const float* ln_q_w = (const float*)d_in[6];
  const float* ln_q_b = (const float*)d_in[7];
  const float* ln_k_w = (const float*)d_in[8];
  const float* ln_k_b = (const float*)d_in[9];
  const float* bias1  = (const float*)d_in[10];
  const float* bias2  = (const float*)d_in[11];
  float* out = (float*)d_out;

  const size_t nW = (size_t)kM * kD * kE; // 2,097,152 elems per weight
  unsigned short* WqT = (unsigned short*)d_ws;
  unsigned short* WkT = WqT + nW;
  unsigned short* WvT = WkT + nW;

  dim3 wg(kE / 32, kD / 32, kM), wb(32, 8);
  convert_w_kernel<<<wg, wb, 0, stream>>>(Wq, WqT);
  convert_w_kernel<<<wg, wb, 0, stream>>>(Wk, WkT);
  convert_w_kernel<<<wg, wb, 0, stream>>>(Wv, WvT);

  fused_agent_attn_kernel<<<dim3(kB / 16, kM), 256, 0, stream>>>(
      q, k, v, WqT, WkT, WvT, ln_q_w, ln_q_b, ln_k_w, ln_k_b, bias1, bias2,
      out);
}
